// Conv2d_28793460752845
// MI455X (gfx1250) — compile-verified, matrix operand order checked by reading, not run on hardware
//
#include <hip/hip_runtime.h>
#include <cstdint>
#include <cstddef>

// ---------------- problem constants ----------------
#define NB   32              // batch
#define CIN  128             // input channels
#define KOUT 256             // output channels
#define HH   56              // height = width
#define HW   (HH*HH)         // 3136
#define HP   58              // padded spatial (1-px halo each side)
#define HP2  (HP*HP)         // 3364
#define NPIX (NB*HW)         // 100352 output pixels
#define WELEM (KOUT*CIN*9)   // 294912 weight elements
#define XPADE ((size_t)NB*HP2*CIN) // 13,778,944 padded input elements

typedef __attribute__((ext_vector_type(16))) __bf16          v16bf;
typedef __attribute__((ext_vector_type(8)))  float           v8f;
typedef __attribute__((ext_vector_type(8)))  unsigned short  u16x8;
typedef __attribute__((ext_vector_type(16))) unsigned short  u16x16;

// ---------------- bf16 split helpers ----------------
__device__ __forceinline__ unsigned short bf16_rne(float x) {
    unsigned u = __builtin_bit_cast(unsigned, x);
    unsigned r = u + 0x7FFFu + ((u >> 16) & 1u);   // round-to-nearest-even
    return (unsigned short)(r >> 16);
}
__device__ __forceinline__ float bf16_f32(unsigned short h) {
    unsigned u = ((unsigned)h) << 16;
    return __builtin_bit_cast(float, u);
}

// ---------------- prologue 1: weights OIHW f32 -> [rs][o][c] bf16 hi/lo ----------------
__global__ __launch_bounds__(256) void repack_weights(const float* __restrict__ k,
                                                      unsigned short* __restrict__ hi,
                                                      unsigned short* __restrict__ lo) {
    int tid = blockIdx.x * 256 + threadIdx.x;
    if (tid >= WELEM) return;
    int o   = tid / (CIN * 9);
    int rem = tid % (CIN * 9);
    int c   = rem / 9;
    int rs  = rem % 9;
    float x = k[tid];                       // OIHW contiguous read
    unsigned short h = bf16_rne(x);
    unsigned short l = bf16_rne(x - bf16_f32(h));
    int oidx = (rs * KOUT + o) * CIN + c;   // c-contiguous (K-major for WMMA A)
    hi[oidx] = h;
    lo[oidx] = l;
}

// ---------------- prologue 2: input NCHW f32 -> padded NHWC bf16 hi/lo ----------------
// Writes the FULL padded array every call (halo = 0), so the main loop needs no
// boundary predication at all.
__global__ __launch_bounds__(256) void repack_input_pad(const float* __restrict__ in,
                                                        unsigned short* __restrict__ hi,
                                                        unsigned short* __restrict__ lo) {
    size_t tid = (size_t)blockIdx.x * 256 + threadIdx.x;  // contiguous over padded NHWC
    if (tid >= XPADE) return;
    int n   = (int)(tid / ((size_t)HP2 * CIN));
    int rem = (int)(tid % ((size_t)HP2 * CIN));
    int pix = rem / CIN;
    int c   = rem % CIN;
    int hh  = pix / HP;
    int ww  = pix % HP;
    int ih  = hh - 1;
    int iw  = ww - 1;
    bool inside = ((unsigned)ih < (unsigned)HH) && ((unsigned)iw < (unsigned)HH);
    float x = inside ? in[(size_t)n * (CIN * HW) + (size_t)c * HW + ih * HH + iw] : 0.0f;
    unsigned short h = bf16_rne(x);
    unsigned short l = bf16_rne(x - bf16_f32(h));
    hi[tid] = h;
    lo[tid] = l;
}

// ---------------- fragment assembly: two 16B chunks -> v16bf ----------------
__device__ __forceinline__ u16x8 ld8(const unsigned short* p) {
    return *(const u16x8*)p;
}
__device__ __forceinline__ v16bf mk_frag(u16x8 a, u16x8 b) {
    u16x16 t = __builtin_shufflevector(a, b, 0,1,2,3,4,5,6,7,8,9,10,11,12,13,14,15);
    return __builtin_bit_cast(v16bf, t);
}
__device__ __forceinline__ v8f wmma_bf16(v16bf a, v16bf b, v8f c) {
    return __builtin_amdgcn_wmma_f32_16x16x32_bf16(false, a, false, b,
                                                   (short)0, c, false, false);
}

// ---------------- main implicit-GEMM conv ----------------
// grid = (NPIX/128, KOUT/64), block = 256 threads = 8 waves (2 in M x 4 in N).
// Each wave: 32x32 output tile = 4 accumulators; 9 taps * 4 c-blocks K-loop;
// per K-step: 16 b128 loads feed 12 v_wmma (hi*hi + hi*lo + lo*hi per acc).
__global__ __launch_bounds__(256) void conv3x3_wmma_bf16x3(
    const unsigned short* __restrict__ Whi, const unsigned short* __restrict__ Wlo,
    const unsigned short* __restrict__ Xhi, const unsigned short* __restrict__ Xlo,
    const float* __restrict__ bias, float* __restrict__ out)
{
    const int lane  = threadIdx.x & 31;
    const int wave  = threadIdx.x >> 5;
    const int mi    = wave >> 2;            // 0..1
    const int ni    = wave & 3;             // 0..3
    const int oBase = blockIdx.y * 64 + mi * 32;
    const int pBase = blockIdx.x * 128 + ni * 32;

    const int lrow  = lane & 15;            // fragment row (M for A, N for B)
    const int khalf = (lane >> 4) * 8;      // K sub-offset per ISA 16-bit layout

    // Two B-side pixels per lane (N-tiles 0 and 1)
    const int p0 = pBase + lrow,      p1 = p0 + 16;
    const int n0 = p0 / HW, hw0 = p0 % HW, h0 = hw0 / HH, w0 = hw0 % HH;
    const int n1 = p1 / HW, hw1 = p1 % HW, h1 = hw1 / HH, w1 = hw1 % HH;
    // padded-NHWC row bases (halo offset +1 folded into (r,s) term below)
    const size_t xb0 = ((size_t)(n0 * HP + h0) * HP + w0) * CIN;
    const size_t xb1 = ((size_t)(n1 * HP + h1) * HP + w1) * CIN;

    // Two A-side output channels per lane (M-tiles 0 and 1)
    const int o0 = oBase + lrow, o1 = o0 + 16;

    v8f a00 = {}, a01 = {}, a10 = {}, a11 = {};   // acc[mtile][ntile]

    for (int rs = 0; rs < 9; ++rs) {
        const int r = rs / 3, s = rs % 3;
        const size_t tap = (size_t)(r * HP + s) * CIN;   // (ih = h+r-1+1, iw = w+s-1+1)
        const unsigned short* xh0 = Xhi + xb0 + tap;
        const unsigned short* xl0 = Xlo + xb0 + tap;
        const unsigned short* xh1 = Xhi + xb1 + tap;
        const unsigned short* xl1 = Xlo + xb1 + tap;
        const size_t wofs = (size_t)rs * (KOUT * CIN);
        const unsigned short* wh0 = Whi + wofs + (size_t)o0 * CIN;
        const unsigned short* wl0 = Wlo + wofs + (size_t)o0 * CIN;
        const unsigned short* wh1 = Whi + wofs + (size_t)o1 * CIN;
        const unsigned short* wl1 = Wlo + wofs + (size_t)o1 * CIN;

        #pragma unroll
        for (int cb = 0; cb < CIN; cb += 32) {
            const int c0 = cb + khalf;
            const int c1 = cb + 16 + khalf;

            // A fragments: 2 M-tiles, hi+lo
            v16bf Ah0 = mk_frag(ld8(wh0 + c0), ld8(wh0 + c1));
            v16bf Al0 = mk_frag(ld8(wl0 + c0), ld8(wl0 + c1));
            v16bf Ah1 = mk_frag(ld8(wh1 + c0), ld8(wh1 + c1));
            v16bf Al1 = mk_frag(ld8(wl1 + c0), ld8(wl1 + c1));
            // B fragments: 2 N-tiles, hi+lo (no masking: halo is zero-padded)
            v16bf Bh0 = mk_frag(ld8(xh0 + c0), ld8(xh0 + c1));
            v16bf Bl0 = mk_frag(ld8(xl0 + c0), ld8(xl0 + c1));
            v16bf Bh1 = mk_frag(ld8(xh1 + c0), ld8(xh1 + c1));
            v16bf Bl1 = mk_frag(ld8(xl1 + c0), ld8(xl1 + c1));

            // x*w ~= wh*xh + wh*xl + wl*xh  (fp32-class accuracy)
            a00 = wmma_bf16(Ah0, Bh0, a00);
            a00 = wmma_bf16(Ah0, Bl0, a00);
            a00 = wmma_bf16(Al0, Bh0, a00);

            a01 = wmma_bf16(Ah0, Bh1, a01);
            a01 = wmma_bf16(Ah0, Bl1, a01);
            a01 = wmma_bf16(Al0, Bh1, a01);

            a10 = wmma_bf16(Ah1, Bh0, a10);
            a10 = wmma_bf16(Ah1, Bl0, a10);
            a10 = wmma_bf16(Al1, Bh0, a10);

            a11 = wmma_bf16(Ah1, Bh1, a11);
            a11 = wmma_bf16(Ah1, Bl1, a11);
            a11 = wmma_bf16(Al1, Bh1, a11);
        }
    }

    // Epilogue: C/D layout -> lane holds N=lane&15; VGPR j is M=j (+8 for lanes>=16)
    const size_t ob0 = (size_t)n0 * (KOUT * HW) + (size_t)hw0;
    const size_t ob1 = (size_t)n1 * (KOUT * HW) + (size_t)hw1;
    {
        const int oR = oBase + khalf;          // M-tile 0
        #pragma unroll
        for (int j = 0; j < 8; ++j) {
            const int oo = oR + j;
            const float bv = bias[oo];
            out[ob0 + (size_t)oo * HW] = a00[j] + bv;
            out[ob1 + (size_t)oo * HW] = a01[j] + bv;
        }
    }
    {
        const int oR = oBase + 16 + khalf;     // M-tile 1
        #pragma unroll
        for (int j = 0; j < 8; ++j) {
            const int oo = oR + j;
            const float bv = bias[oo];
            out[ob0 + (size_t)oo * HW] = a10[j] + bv;
            out[ob1 + (size_t)oo * HW] = a11[j] + bv;
        }
    }
}

// ---------------- host-side launch ----------------
extern "C" void kernel_launch(void* const* d_in, const int* in_sizes, int n_in,
                              void* d_out, int out_size, void* d_ws, size_t ws_size,
                              hipStream_t stream) {
    const float* inp  = (const float*)d_in[0];   // (32,128,56,56) f32
    const float* ker  = (const float*)d_in[1];   // (256,128,3,3)  f32
    const float* bias = (const float*)d_in[2];   // (256,)         f32
    float* out = (float*)d_out;                  // (32,256,56,56) f32

    unsigned short* Whi = (unsigned short*)d_ws;
    unsigned short* Wlo = Whi + (size_t)WELEM;
    unsigned short* Xhi = Wlo + (size_t)WELEM;
    unsigned short* Xlo = Xhi + XPADE;
    // total ws use: 2*294912*2B + 2*13778944*2B ~= 56.3 MB

    repack_weights<<<(WELEM + 255) / 256, 256, 0, stream>>>(ker, Whi, Wlo);
    repack_input_pad<<<(unsigned)((XPADE + 255) / 256), 256, 0, stream>>>(inp, Xhi, Xlo);

    dim3 grid(NPIX / 128, KOUT / 64);   // (784, 4)
    conv3x3_wmma_bf16x3<<<grid, 256, 0, stream>>>(Whi, Wlo, Xhi, Xlo, bias, out);
}
